// SEDDCond_45208825758308
// MI455X (gfx1250) — matrix operand order repaired and checked by reading, not compiled, not run on hardware
//
#include <hip/hip_runtime.h>

typedef __attribute__((ext_vector_type(16))) _Float16 v16h;
typedef __attribute__((ext_vector_type(8)))  _Float16 v8h;
typedef __attribute__((ext_vector_type(8)))  float    v8f;

#define S_TOT   32768      // 32^3 tokens per batch
#define D_MODEL 512
#define NSEL    512        // selected blocks (both batches)
#define NROWS   (NSEL*64)  // 32768 selected token rows
#define MROWS   (2*S_TOT)  // 65536 total token rows

__device__ __forceinline__ v16h ldfrag(const _Float16* p, int off) {
  union { v8h h[2]; v16h v; } u;
  u.h[0] = *(const v8h*)(p);
  u.h[1] = *(const v8h*)(p + off);
  return u.v;
}

__device__ __forceinline__ v8f wmma16(v16h a, v16h b, v8f c) {
  return __builtin_amdgcn_wmma_f32_16x16x32_f16(false, a, false, b, (short)0, c,
                                                false, false);
}

// ---- CDNA5 async global->LDS (ASYNCcnt-tracked DMA, bypasses VGPRs) --------
__device__ __forceinline__ unsigned lds_off(const void* p) {
  return (unsigned)(size_t)p;        // low 32 bits of generic ptr = LDS offset
}
__device__ __forceinline__ void async_b128(unsigned lds, const void* g) {
  asm volatile("global_load_async_to_lds_b128 %0, %1, off"
               :: "v"(lds), "v"((unsigned long long)(size_t)g) : "memory");
}
__device__ __forceinline__ void wait_async_le4() {
  asm volatile("s_wait_asynccnt 0x4" ::: "memory");
}
__device__ __forceinline__ void wait_async_0() {
  asm volatile("s_wait_asynccnt 0x0" ::: "memory");
}

// ---------------------------------------------------------------- f32 -> f16
__global__ __launch_bounds__(256) void k_cvt(const float* __restrict__ in,
                                             _Float16* __restrict__ out, int n) {
  int i = blockIdx.x * 256 + threadIdx.x;
  if (i < n) out[i] = (_Float16)in[i];
}

// ------------------------------------------------- adaLN: mods = c@Wt + b
__global__ __launch_bounds__(256) void k_adaln(const float* __restrict__ c,
                                               const float* __restrict__ w,
                                               const float* __restrict__ bias,
                                               float* __restrict__ mods) {
  int j = blockIdx.x * 256 + threadIdx.x;      // 0..6143
  int b = j / 3072, o = j % 3072;
  const float4* cv = (const float4*)(c + (size_t)b * 512);
  const float4* wv = (const float4*)(w + (size_t)o * 512);
  float acc = 0.f;
  for (int k = 0; k < 128; ++k) {
    float4 a = cv[k], d = wv[k];
    acc += a.x * d.x + a.y * d.y + a.z * d.z + a.w * d.w;
  }
  mods[j] = acc + bias[o];
}

// ------------------------------------------------- block-id -> selected rank
__global__ void k_rank(const int* __restrict__ nz, int* __restrict__ rank) {
  int t = threadIdx.x;                 // blockDim = 1024
  rank[t] = -1;
  __syncthreads();
  if (t < NSEL) rank[nz[t]] = t;
}

// --------------------------------------- copy x -> x1 (residual base, f32)
__global__ __launch_bounds__(256) void k_copy4(const float4* __restrict__ in,
                                               float4* __restrict__ out) {
  size_t i = (size_t)blockIdx.x * 256 + threadIdx.x;
  out[i] = in[i];
}

// ---------- LN1 + modulate + gather selected blocks -> f16 A-matrix + rowmap
__global__ __launch_bounds__(256)
void k_ln1(const float* __restrict__ x, const float* __restrict__ w1,
           const float* __restrict__ mods, const int* __restrict__ nz,
           _Float16* __restrict__ A, int* __restrict__ row_map) {
  int lane = threadIdx.x & 31, wid = threadIdx.x >> 5;
  int r = blockIdx.x * 8 + wid;                 // selected row 0..32767
  int i = r >> 6, p = r & 63;
  int blk = nz[i];
  int b = blk >> 9, rem = blk & 511;
  int bx = rem >> 6, by = (rem >> 3) & 7, bz = rem & 7;
  int px = p >> 4, py = (p >> 2) & 3, pz = p & 3;
  int s = ((bx * 4 + px) * 32 + (by * 4 + py)) * 32 + (bz * 4 + pz);
  const float* row = x + ((size_t)b * S_TOT + s) * D_MODEL;

  float vals[16];
  const float4* rp = (const float4*)row;
#pragma unroll
  for (int q = 0; q < 4; ++q) {
    float4 t = rp[lane * 4 + q];
    vals[q * 4 + 0] = t.x; vals[q * 4 + 1] = t.y;
    vals[q * 4 + 2] = t.z; vals[q * 4 + 3] = t.w;
  }
  float sum = 0.f;
#pragma unroll
  for (int e = 0; e < 16; ++e) sum += vals[e];
  for (int o = 16; o > 0; o >>= 1) sum += __shfl_xor(sum, o, 32);
  float mu = sum * (1.f / 512.f);
  float vs = 0.f;
#pragma unroll
  for (int e = 0; e < 16; ++e) { float d = vals[e] - mu; vs += d * d; }
  for (int o = 16; o > 0; o >>= 1) vs += __shfl_xor(vs, o, 32);
  float rstd = rsqrtf(vs * (1.f / 512.f) + 1e-5f);

  const float* md = mods + (size_t)b * 3072;
  union { _Float16 h[16]; uint4 u[2]; } o16;
#pragma unroll
  for (int e = 0; e < 16; ++e) {
    int col = lane * 16 + e;
    float y = (vals[e] - mu) * rstd * w1[col];
    y = y * (1.f + md[512 + col]) + md[col];
    o16.h[e] = (_Float16)y;
  }
  uint4* dst = (uint4*)(A + (size_t)r * D_MODEL + lane * 16);
  dst[0] = o16.u[0]; dst[1] = o16.u[1];
  if (lane == 0) row_map[r] = b * S_TOT + s;
}

// ---------------- LN2 + modulate over all rows of x1 -> f16 MLP A-matrix
__global__ __launch_bounds__(256)
void k_ln2(const float* __restrict__ x1, const float* __restrict__ w2,
           const float* __restrict__ mods, _Float16* __restrict__ A) {
  int lane = threadIdx.x & 31, wid = threadIdx.x >> 5;
  int r = blockIdx.x * 8 + wid;                 // 0..65535
  int b = r >> 15;
  const float* row = x1 + (size_t)r * D_MODEL;

  float vals[16];
  const float4* rp = (const float4*)row;
#pragma unroll
  for (int q = 0; q < 4; ++q) {
    float4 t = rp[lane * 4 + q];
    vals[q * 4 + 0] = t.x; vals[q * 4 + 1] = t.y;
    vals[q * 4 + 2] = t.z; vals[q * 4 + 3] = t.w;
  }
  float sum = 0.f;
#pragma unroll
  for (int e = 0; e < 16; ++e) sum += vals[e];
  for (int o = 16; o > 0; o >>= 1) sum += __shfl_xor(sum, o, 32);
  float mu = sum * (1.f / 512.f);
  float vs = 0.f;
#pragma unroll
  for (int e = 0; e < 16; ++e) { float d = vals[e] - mu; vs += d * d; }
  for (int o = 16; o > 0; o >>= 1) vs += __shfl_xor(vs, o, 32);
  float rstd = rsqrtf(vs * (1.f / 512.f) + 1e-5f);

  const float* md = mods + (size_t)b * 3072;
  union { _Float16 h[16]; uint4 u[2]; } o16;
#pragma unroll
  for (int e = 0; e < 16; ++e) {
    int col = lane * 16 + e;
    float y = (vals[e] - mu) * rstd * w2[col];
    y = y * (1.f + md[2048 + col]) + md[1536 + col];
    o16.h[e] = (_Float16)y;
  }
  uint4* dst = (uint4*)(A + (size_t)r * D_MODEL + lane * 16);
  dst[0] = o16.u[0]; dst[1] = o16.u[1];
}

// ---------------- RoPE in-place on Q,K of qkv buffer + roped empty K / V
__global__ __launch_bounds__(256)
void k_rope(_Float16* __restrict__ qkv, const float* __restrict__ cq,
            const float* __restrict__ sq, const float* __restrict__ ck,
            const float* __restrict__ sk, const float* __restrict__ kv_empty,
            _Float16* __restrict__ keh, _Float16* __restrict__ veh) {
  int r = blockIdx.x;
  int hh = threadIdx.x >> 5, d = threadIdx.x & 31;
  if (r < NROWS) {
    int p = r & 63;
    _Float16* q = qkv + (size_t)r * 1536 + hh * 64 + d;
    float c = cq[p * 64 + d], s = sq[p * 64 + d];
    float a0 = (float)q[0], a1 = (float)q[32];
    q[0]  = (_Float16)(a0 * c - a1 * s);
    q[32] = (_Float16)(a1 * c + a0 * s);
    _Float16* k = q + 512;
    c = ck[p * 64 + d]; s = sk[p * 64 + d];
    a0 = (float)k[0]; a1 = (float)k[32];
    k[0]  = (_Float16)(a0 * c - a1 * s);
    k[32] = (_Float16)(a1 * c + a0 * s);
  } else {
    int p = r - NROWS;                       // 0..63
    const float* ke = kv_empty + (size_t)p * 1024;
    float c = ck[p * 64 + d], s = sk[p * 64 + d];
    float a0 = ke[hh * 64 + d], a1 = ke[hh * 64 + d + 32];
    keh[p * 512 + hh * 64 + d]      = (_Float16)(a0 * c - a1 * s);
    keh[p * 512 + hh * 64 + d + 32] = (_Float16)(a1 * c + a0 * s);
    veh[p * 512 + hh * 64 + d]      = (_Float16)ke[512 + hh * 64 + d];
    veh[p * 512 + hh * 64 + d + 32] = (_Float16)ke[512 + hh * 64 + d + 32];
  }
}

// ------------------------- WMMA GEMM: C[M,N] = A[M,K](f16) * W[N,K](f16)^T
// Triple-buffered LDS ring fed by async global->LDS DMA; one barrier per step.
// epi 0: store f16            epi 1: +bias, gelu, store f16
// epi 2: x1[row_map] += gate_msa * v      epi 3: out = x1 + gate_mlp*(v+bias)
__global__ __launch_bounds__(256)
void k_gemm(const _Float16* __restrict__ A, const _Float16* __restrict__ W,
            int M, int N, int K, int epi,
            _Float16* __restrict__ outH, float* __restrict__ outF,
            const float* __restrict__ bias, float* __restrict__ x1,
            const float* __restrict__ mods, const int* __restrict__ row_map) {
  __shared__ _Float16 sA[3][128][40];
  __shared__ _Float16 sB[3][128][40];
  const int tid = threadIdx.x;
  const int lane = tid & 31, wid = tid >> 5;
  const int wm = wid & 3;        // 4 M-strips of 32
  const int wn = wid >> 2;       // 2 N-strips of 64
  const int bm = blockIdx.x, bn = blockIdx.y;
  const int lr = tid >> 1;             // 0..127
  const int lc = (tid & 1) * 16;       // 0 | 16

  const _Float16* gA = A + (size_t)(bm * 128 + lr) * K + lc;
  const _Float16* gW = W + (size_t)(bn * 128 + lr) * K + lc;
  const unsigned ldsA0 = lds_off(&sA[0][lr][lc]);
  const unsigned ldsB0 = lds_off(&sB[0][lr][lc]);
  const unsigned BSTR = 128u * 40u * 2u;      // bytes per ring slot

  const int nsteps = K >> 5;
  // prologue: fill slots 0 and 1
  async_b128(ldsA0,      gA);      async_b128(ldsA0 + 16, gA + 8);
  async_b128(ldsB0,      gW);      async_b128(ldsB0 + 16, gW + 8);
  async_b128(ldsA0 + BSTR,      gA + 32); async_b128(ldsA0 + BSTR + 16, gA + 40);
  async_b128(ldsB0 + BSTR,      gW + 32); async_b128(ldsB0 + BSTR + 16, gW + 40);

  v8f acc[2][4];
#pragma unroll
  for (int mi = 0; mi < 2; ++mi)
#pragma unroll
    for (int ni = 0; ni < 4; ++ni)
#pragma unroll
      for (int e = 0; e < 8; ++e) acc[mi][ni][e] = 0.f;

  int buf = 0;
  for (int s = 0; s < nsteps; ++s) {
    if (s + 1 < nsteps) wait_async_le4();   // slot s landed, slot s+1 in flight
    else                wait_async_0();
    __syncthreads();                        // everyone's slot-s DMA complete,
                                            // everyone done reading slot used at s-1
    const _Float16* cA = &sA[buf][0][0];
    const _Float16* cB = &sB[buf][0][0];
    v16h af[2], bf[4];
#pragma unroll
    for (int mi = 0; mi < 2; ++mi)
      af[mi] = ldfrag(cA + (wm * 32 + mi * 16 + (lane & 15)) * 40 +
                      ((lane & 16) ? 8 : 0), 16);
#pragma unroll
    for (int ni = 0; ni < 4; ++ni)
      bf[ni] = ldfrag(cB + (wn * 64 + ni * 16 + (lane & 15)) * 40 +
                      ((lane & 16) ? 8 : 0), 16);
#pragma unroll
    for (int mi = 0; mi < 2; ++mi)
#pragma unroll
      for (int ni = 0; ni < 4; ++ni) acc[mi][ni] = wmma16(af[mi], bf[ni], acc[mi][ni]);

    if (s + 2 < nsteps) {                   // refill slot read at step s-1
      int nb = buf + 2; if (nb >= 3) nb -= 3;
      const _Float16* nA = gA + (size_t)(s + 2) * 32;
      const _Float16* nW = gW + (size_t)(s + 2) * 32;
      unsigned la = ldsA0 + (unsigned)nb * BSTR;
      unsigned lb = ldsB0 + (unsigned)nb * BSTR;
      async_b128(la, nA); async_b128(la + 16, nA + 8);
      async_b128(lb, nW); async_b128(lb + 16, nW + 8);
    }
    if (++buf == 3) buf = 0;
  }

#pragma unroll
  for (int mi = 0; mi < 2; ++mi)
#pragma unroll
    for (int ni = 0; ni < 4; ++ni)
#pragma unroll
      for (int e = 0; e < 8; ++e) {
        int row = bm * 128 + wm * 32 + mi * 16 + e + ((lane & 16) ? 8 : 0);
        int col = bn * 128 + wn * 64 + ni * 16 + (lane & 15);
        float v = acc[mi][ni][e];
        if (epi == 0) {
          outH[(size_t)row * N + col] = (_Float16)v;
        } else if (epi == 1) {
          v += bias[col];
          float g = 0.5f * v * (1.f + tanhf(0.7978845608f * (v + 0.044715f * v * v * v)));
          outH[(size_t)row * N + col] = (_Float16)g;
        } else if (epi == 2) {
          int idx = row_map[row];
          int b = idx >> 15;
          float g = mods[(size_t)b * 3072 + 1024 + col];
          size_t o = (size_t)idx * D_MODEL + col;
          x1[o] = x1[o] + g * v;
        } else {
          v += bias[col];
          int b = row >> 15;
          float g = mods[(size_t)b * 3072 + 2560 + col];
          size_t o = (size_t)row * D_MODEL + col;
          outF[o] = x1[o] + g * v;
        }
      }
}

// ------------------- block-sparse neighbor attention, one WG per (block,head)
__global__ __launch_bounds__(128)
void k_attn(const _Float16* __restrict__ qkv, const _Float16* __restrict__ keh,
            const _Float16* __restrict__ veh, const int* __restrict__ rank,
            const int* __restrict__ nbr, _Float16* __restrict__ attn_out) {
  extern __shared__ unsigned char smem[];
  _Float16* sQ  = (_Float16*)smem;          // [64][72]
  _Float16* sKV = sQ + 64 * 72;             // [64][72] (K tile, later V^T tile)
  float*    sS  = (float*)(sKV + 64 * 72);  // [64][456]
  _Float16* sP  = (_Float16*)(sS + 64 * 456); // [64][456]

  const int i  = blockIdx.x;      // selected block
  const int hh = blockIdx.y;      // head
  const int tid = threadIdx.x;
  const int lane = tid & 31, w = tid >> 5;   // wave = 16-query strip

  { // stage Q head-slice (64x64) via async DMA
    int q = tid >> 1, d0 = (tid & 1) * 32;
    const _Float16* src = qkv + (size_t)(i * 64 + q) * 1536 + hh * 64 + d0;
    unsigned dst = lds_off(sQ + q * 72 + d0);
#pragma unroll
    for (int t = 0; t < 4; ++t) async_b128(dst + 16 * t, src + 8 * t);
  }
  wait_async_0();
  __syncthreads();

  // phase 1: scores = Q K^T per neighbor
  for (int j = 0; j < 7; ++j) {
    int g = nbr[i * 7 + j];
    int r = rank[g];
    {
      int ky = tid >> 1, d0 = (tid & 1) * 32;
      const _Float16* src = (r >= 0)
          ? qkv + (size_t)((size_t)r * 64 + ky) * 1536 + 512 + hh * 64 + d0
          : keh + (size_t)ky * 512 + hh * 64 + d0;
      unsigned dst = lds_off(sKV + ky * 72 + d0);
#pragma unroll
      for (int t = 0; t < 4; ++t) async_b128(dst + 16 * t, src + 8 * t);
    }
    wait_async_0();
    __syncthreads();

    v8f sacc[4];
#pragma unroll
    for (int ni = 0; ni < 4; ++ni)
#pragma unroll
      for (int e = 0; e < 8; ++e) sacc[ni][e] = 0.f;
#pragma unroll
    for (int ks = 0; ks < 2; ++ks) {
      v16h a = ldfrag(&sQ[(w * 16 + (lane & 15)) * 72 + ks * 32 + ((lane & 16) ? 8 : 0)], 16);
#pragma unroll
      for (int ni = 0; ni < 4; ++ni) {
        v16h b = ldfrag(&sKV[(ni * 16 + (lane & 15)) * 72 + ks * 32 + ((lane & 16) ? 8 : 0)], 16);
        sacc[ni] = wmma16(a, b, sacc[ni]);
      }
    }
#pragma unroll
    for (int ni = 0; ni < 4; ++ni)
#pragma unroll
      for (int e = 0; e < 8; ++e) {
        int qr = w * 16 + e + ((lane & 16) ? 8 : 0);
        sS[qr * 456 + j * 64 + ni * 16 + (lane & 15)] = sacc[ni][e];
      }
    __syncthreads();
  }

  // softmax rows (scale 1/sqrt(64) folded in)
  if (tid < 64) {
    int q = tid;
    float m = -1e30f;
    for (int c = 0; c < 448; ++c) m = fmaxf(m, sS[q * 456 + c] * 0.125f);
    float sum = 0.f;
    for (int c = 0; c < 448; ++c) {
      float e = __expf(sS[q * 456 + c] * 0.125f - m);
      sum += e;
      sP[q * 456 + c] = (_Float16)e;
    }
    float inv = 1.f / sum;
    for (int c = 0; c < 448; ++c)
      sP[q * 456 + c] = (_Float16)((float)sP[q * 456 + c] * inv);
  }
  __syncthreads();

  // phase 2: O = P V (V staged transposed so B-frag is key-major)
  v8f oacc[4];
#pragma unroll
  for (int ni = 0; ni < 4; ++ni)
#pragma unroll
    for (int e = 0; e < 8; ++e) oacc[ni][e] = 0.f;

  for (int j = 0; j < 7; ++j) {
    int g = nbr[i * 7 + j];
    int r = rank[g];
    {
      int ky = tid >> 1, d0 = (tid & 1) * 32;
      const _Float16* src = (r >= 0)
          ? qkv + (size_t)((size_t)r * 64 + ky) * 1536 + 1024 + hh * 64 + d0
          : veh + (size_t)ky * 512 + hh * 64 + d0;
      union { uint4 u[4]; _Float16 h[32]; } tv;
      const uint4* s4 = (const uint4*)src;
      tv.u[0] = s4[0]; tv.u[1] = s4[1]; tv.u[2] = s4[2]; tv.u[3] = s4[3];
#pragma unroll
      for (int d = 0; d < 32; ++d) sKV[(d0 + d) * 72 + ky] = tv.h[d];  // V^T
    }
    __syncthreads();
#pragma unroll
    for (int ks = 0; ks < 2; ++ks) {
      v16h a = ldfrag(&sP[(w * 16 + (lane & 15)) * 456 + j * 64 + ks * 32 +
                          ((lane & 16) ? 8 : 0)], 16);
#pragma unroll
      for (int ni = 0; ni < 4; ++ni) {
        v16h b = ldfrag(&sKV[(ni * 16 + (lane & 15)) * 72 + ks * 32 +
                             ((lane & 16) ? 8 : 0)], 16);
        oacc[ni] = wmma16(a, b, oacc[ni]);
      }
    }
    __syncthreads();
  }

#pragma unroll
  for (int ni = 0; ni < 4; ++ni)
#pragma unroll
    for (int e = 0; e < 8; ++e) {
      int qr = w * 16 + e + ((lane & 16) ? 8 : 0);
      int d  = ni * 16 + (lane & 15);
      attn_out[(size_t)(i * 64 + qr) * D_MODEL + hh * 64 + d] = (_Float16)oacc[ni][e];
    }
}

// ============================================================== host launcher
extern "C" void kernel_launch(void* const* d_in, const int* in_sizes, int n_in,
                              void* d_out, int out_size, void* d_ws, size_t ws_size,
                              hipStream_t stream) {
  const float* x        = (const float*)d_in[0];
  const float* c        = (const float*)d_in[1];
  const float* cos_q    = (const float*)d_in[2];
  const float* sin_q    = (const float*)d_in[3];
  const float* cos_k    = (const float*)d_in[4];
  const float* sin_k    = (const float*)d_in[5];
  const int*   nz       = (const int*)d_in[6];
  const int*   nbrs     = (const int*)d_in[7];
  const float* norm1_w  = (const float*)d_in[11];
  const float* qkv_w    = (const float*)d_in[12];
  const float* out_w    = (const float*)d_in[13];
  const float* norm2_w  = (const float*)d_in[14];
  const float* mlp_w1   = (const float*)d_in[15];
  const float* mlp_b1   = (const float*)d_in[16];
  const float* mlp_w2   = (const float*)d_in[17];
  const float* mlp_b2   = (const float*)d_in[18];
  const float* adaln_w  = (const float*)d_in[19];
  const float* adaln_b  = (const float*)d_in[20];
  const float* kv_empty = (const float*)d_in[21];
  float* out = (float*)d_out;

  // workspace carve-up (256B aligned)
  size_t off = 0;
  char* base = (char*)d_ws;
  auto carve = [&](size_t bytes) -> void* {
    void* p = base + off;
    off += (bytes + 255) & ~(size_t)255;
    return p;
  };
  _Float16* wQKV  = (_Float16*)carve((size_t)1536 * 512 * 2);
  _Float16* wOUT  = (_Float16*)carve((size_t)512 * 512 * 2);
  _Float16* wM1   = (_Float16*)carve((size_t)2048 * 512 * 2);
  _Float16* wM2   = (_Float16*)carve((size_t)512 * 2048 * 2);
  float*    mods  = (float*)carve((size_t)2 * 3072 * 4);
  int*      rank  = (int*)carve((size_t)1024 * 4);
  int*      rmap  = (int*)carve((size_t)NROWS * 4);
  _Float16* keh   = (_Float16*)carve((size_t)64 * 512 * 2);
  _Float16* veh   = (_Float16*)carve((size_t)64 * 512 * 2);
  _Float16* Aqkv  = (_Float16*)carve((size_t)NROWS * 512 * 2);
  _Float16* QKV   = (_Float16*)carve((size_t)NROWS * 1536 * 2);
  _Float16* AttnO = (_Float16*)carve((size_t)NROWS * 512 * 2);
  float*    x1    = (float*)carve((size_t)MROWS * 512 * 4);
  _Float16* Amlp  = (_Float16*)carve((size_t)MROWS * 512 * 2);
  _Float16* Hh    = (_Float16*)carve((size_t)MROWS * 2048 * 2);
  (void)ws_size; (void)in_sizes; (void)n_in; (void)out_size;

  // 1. weight casts
  k_cvt<<<(1536 * 512 + 255) / 256, 256, 0, stream>>>(qkv_w, wQKV, 1536 * 512);
  k_cvt<<<(512 * 512 + 255) / 256, 256, 0, stream>>>(out_w, wOUT, 512 * 512);
  k_cvt<<<(2048 * 512 + 255) / 256, 256, 0, stream>>>(mlp_w1, wM1, 2048 * 512);
  k_cvt<<<(512 * 2048 + 255) / 256, 256, 0, stream>>>(mlp_w2, wM2, 512 * 2048);

  // 2. adaLN modulation vectors, block-rank table
  k_adaln<<<(2 * 3072) / 256, 256, 0, stream>>>(c, adaln_w, adaln_b, mods);
  k_rank<<<1, 1024, 0, stream>>>(nz, rank);

  // 3. LN1 + modulate + gather -> f16 A-matrix; QKV GEMM; RoPE
  k_ln1<<<NROWS / 8, 256, 0, stream>>>(x, norm1_w, mods, nz, Aqkv, rmap);
  k_gemm<<<dim3(NROWS / 128, 1536 / 128), 256, 0, stream>>>(
      Aqkv, wQKV, NROWS, 1536, 512, 0, QKV, nullptr, nullptr, nullptr, nullptr, nullptr);
  k_rope<<<NROWS + 64, 256, 0, stream>>>(QKV, cos_q, sin_q, cos_k, sin_k,
                                         kv_empty, keh, veh);

  // 4. residual base x1 = x; attention; gated out-proj scatter into x1
  k_copy4<<<(MROWS * 512 / 4) / 256, 256, 0, stream>>>((const float4*)x, (float4*)x1);
  size_t smem = (size_t)64 * 72 * 2 * 2 + (size_t)64 * 456 * 4 + (size_t)64 * 456 * 2;
  k_attn<<<dim3(NSEL, 8), 128, smem, stream>>>(QKV, keh, veh, rank, nbrs, AttnO);
  k_gemm<<<dim3(NROWS / 128, 512 / 128), 256, 0, stream>>>(
      AttnO, wOUT, NROWS, 512, 512, 2, nullptr, nullptr, nullptr, x1, mods, rmap);

  // 5. LN2 + modulate; MLP with fused gelu / gated residual epilogues
  k_ln2<<<MROWS / 8, 256, 0, stream>>>(x1, norm2_w, mods, Amlp);
  k_gemm<<<dim3(MROWS / 128, 2048 / 128), 256, 0, stream>>>(
      Amlp, wM1, MROWS, 2048, 512, 1, Hh, nullptr, mlp_b1, nullptr, nullptr, nullptr);
  k_gemm<<<dim3(MROWS / 128, 512 / 128), 256, 0, stream>>>(
      Hh, wM2, MROWS, 512, 2048, 3, nullptr, out, mlp_b2, x1, mods, nullptr);
}
